// DifferentialDualStreamAttention_24567212933933
// MI455X (gfx1250) — matrix operand order repaired and checked by reading, not compiled, not run on hardware
//
#include <hip/hip_runtime.h>
#include <hip/hip_bf16.h>

typedef __attribute__((ext_vector_type(16))) _Float16 v16h;
typedef __attribute__((ext_vector_type(8)))  _Float16 h8;
typedef __attribute__((ext_vector_type(8)))  float    v8f;
typedef __attribute__((ext_vector_type(4)))  float    f4;

#define NSEQ 49
#define CDIM 192
#define HEADS 6
#define HDIM 32

__device__ __forceinline__ v8f vz8() {
    v8f z;
#pragma unroll
    for (int i = 0; i < 8; ++i) z[i] = 0.0f;
    return z;
}

__device__ __forceinline__ h8 hz8() {
    h8 z;
#pragma unroll
    for (int i = 0; i < 8; ++i) z[i] = (_Float16)0.0f;
    return z;
}

__device__ __forceinline__ v16h cat16(h8 lo, h8 hi) {
    return __builtin_shufflevector(lo, hi, 0, 1, 2, 3, 4, 5, 6, 7, 8, 9, 10, 11, 12, 13, 14, 15);
}

__device__ __forceinline__ v8f wmma_f16(v16h a, v16h b, v8f c) {
    // emits v_wmma_f32_16x16x32_f16
    return __builtin_amdgcn_wmma_f32_16x16x32_f16(false, a, false, b, (short)0, c, false, false);
}

// row remap for the CA (B, 2N, C) interleave:
// 0: identity; 1: t rows (b*2N + n); 2: r rows (b*2N + N + n)   [m = b*N + n]
__device__ __forceinline__ int map_row(int m, int mp) {
    if (mp == 0) return m;
    return (2 * (m / NSEQ) + (mp - 1)) * NSEQ + (m % NSEQ);
}

// ---------------------------------------------------------------------------
// Small helper kernels
// ---------------------------------------------------------------------------
__global__ void cvt_f32_f16(const float* __restrict__ s, _Float16* __restrict__ d, long long n) {
    long long i = (long long)blockIdx.x * blockDim.x + threadIdx.x;
    long long stride = (long long)gridDim.x * blockDim.x;
    for (; i < n; i += stride) d[i] = (_Float16)s[i];
}

// bias[h][i][j] = rpb_table[ ((ih_i-ih_j+6)*13 + (iw_i-iw_j+6)) ][h]
__global__ void build_bias_kernel(const float* __restrict__ rpb, float* __restrict__ bias) {
    int idx = blockIdx.x * blockDim.x + threadIdx.x;
    if (idx >= HEADS * NSEQ * NSEQ) return;
    int h = idx / (NSEQ * NSEQ);
    int rem = idx % (NSEQ * NSEQ);
    int i = rem / NSEQ, j = rem % NSEQ;
    int d0 = (i / 7 - j / 7) + 6;
    int d1 = (i % 7 - j % 7) + 6;
    bias[idx] = rpb[(d0 * 13 + d1) * HEADS + h];
}

// ---------------------------------------------------------------------------
// Generic WMMA GEMM:  out = alpha * (A @ Bw) + bias + addend
//   A: f16 (Ah) or f32 (Af, converted in-register while staging LDS);
//     a_map remaps A rows for the CA interleaved input.
//   addend: f16 (addH) or f32 (addF) with add_map row remap.
//   outputs: f16 (outH) and/or f32 (outF, with out_map row remap).
// B tile staged in WMMA-B fragment order [cTile][lane][j] (pad 24h/48B) so
// fragments are ds_load_b128 pairs. Block: 128 thr = 4 waves; tile 64x64; K step 32.
// ---------------------------------------------------------------------------
__global__ __launch_bounds__(128) void gemm_f16_wmma(
    const _Float16* __restrict__ Ah, const float* __restrict__ Af, int lda, int a_map,
    const _Float16* __restrict__ Bw, int ldb,
    const float* __restrict__ bias,
    const _Float16* __restrict__ addH, const float* __restrict__ addF, int ldadd, int add_map,
    const float* __restrict__ alpha_ptr,
    _Float16* __restrict__ outH, int ldoH,
    float* __restrict__ outF, int ldoF, int out_map,
    int M, int Nout, int K)
{
    __shared__ _Float16 As[64][40];        // row-major A tile (A-frag chunks contiguous)
    __shared__ _Float16 BsSw[4][32][24];   // B tile in fragment order

    const int tid  = threadIdx.x;
    const int lane = tid & 31;
    const int w    = tid >> 5;
    const int l16  = lane & 15;
    const int hi   = lane >> 4;
    const int kb   = hi * 8;
    const int m0   = blockIdx.y * 64;
    const int n0   = blockIdx.x * 64;

    v8f acc[4];
#pragma unroll
    for (int c = 0; c < 4; ++c) acc[c] = vz8();

    const int arow = tid >> 1, aseg = tid & 1;   // A tile: 64 rows x 2 segs(16 elems)
    const int brow = tid >> 2, bseg = tid & 3;   // B tile: 32 rows x 4 segs(16h)
    const int bhi  = (brow >> 4) * 16, bj = brow & 15;
    const bool aOK  = (m0 + arow) < M;
    const int  asrc = aOK ? map_row(m0 + arow, a_map) : 0;   // hoisted: row-invariant over K

    for (int k0 = 0; k0 < K; k0 += 32) {
        {   // stage A tile
            h8* dst = (h8*)&As[arow][aseg * 16];
            if (!aOK) {
                dst[0] = hz8(); dst[1] = hz8();
            } else if (Ah) {
                const _Float16* src = Ah + (size_t)asrc * lda + k0 + aseg * 16;
                dst[0] = ((const h8*)src)[0];
                dst[1] = ((const h8*)src)[1];
                if (k0 + 32 < K) __builtin_prefetch(src + 32, 0, 1);  // global_prefetch_b8
            } else {
                const float* src = Af + (size_t)asrc * lda + k0 + aseg * 16;
                f4 x0 = ((const f4*)src)[0];
                f4 x1 = ((const f4*)src)[1];
                f4 x2 = ((const f4*)src)[2];
                f4 x3 = ((const f4*)src)[3];
                h8 lo, hh;
#pragma unroll
                for (int i = 0; i < 4; ++i) {
                    lo[i]     = (_Float16)x0[i];
                    lo[4 + i] = (_Float16)x1[i];
                    hh[i]     = (_Float16)x2[i];
                    hh[4 + i] = (_Float16)x3[i];
                }
                dst[0] = lo; dst[1] = hh;
                if (k0 + 32 < K) __builtin_prefetch(src + 32, 0, 1);
            }
        }
        {   // stage B tile -> fragment-order swizzle (vector global load, scalar LDS store)
            const _Float16* src = Bw + (size_t)(k0 + brow) * ldb + n0 + bseg * 16;
            h8 v0 = ((const h8*)src)[0];
            h8 v1 = ((const h8*)src)[1];
            _Float16* dst = &BsSw[bseg][bhi][bj];
#pragma unroll
            for (int i = 0; i < 8; ++i) {
                dst[i * 24]       = v0[i];
                dst[(8 + i) * 24] = v1[i];
            }
        }
        __syncthreads();

        v16h a = cat16(*(const h8*)&As[16 * w + l16][hi * 16],
                       *(const h8*)&As[16 * w + l16][hi * 16 + 8]);
#pragma unroll
        for (int c = 0; c < 4; ++c) {
            h8 b0 = *(const h8*)&BsSw[c][lane][0];
            h8 b1 = *(const h8*)&BsSw[c][lane][8];
            acc[c] = wmma_f16(a, cat16(b0, b1), acc[c]);
        }
        __syncthreads();
    }

    const float alpha = alpha_ptr ? *alpha_ptr : 1.0f;
#pragma unroll
    for (int r = 0; r < 8; ++r) {
        const int gm = m0 + 16 * w + hi * 8 + r;
        if (gm >= M) continue;
        const size_t addoff = (addH || addF) ? (size_t)map_row(gm, add_map) * ldadd : 0;
        const size_t ooff   = outF ? (size_t)map_row(gm, out_map) * ldoF : 0;
        const size_t hoff2  = (size_t)gm * ldoH;
#pragma unroll
        for (int c = 0; c < 4; ++c) {
            const int gn = n0 + c * 16 + l16;
            float v = alpha * acc[c][r];
            if (bias) v += bias[gn];
            if (addH) v += (float)addH[addoff + gn];
            if (addF) v += addF[addoff + gn];
            if (outH) outH[hoff2 + gn] = (_Float16)v;
            if (outF) outF[ooff + gn] = v;
        }
    }
}

// ---------------------------------------------------------------------------
// Fused differential attention, one block per (batch, head).
// K tiles in LDS pre-swizzled to B-fragment order (staging itself vectorized);
// V tiles swizzled with scalar stores (once per block); PV fragments are
// ds_load_b128 pairs.
// ---------------------------------------------------------------------------
__global__ __launch_bounds__(128) void diff_attn_kernel(
    const _Float16* __restrict__ qT, const _Float16* __restrict__ kT, const _Float16* __restrict__ vT,
    const _Float16* __restrict__ qR, const _Float16* __restrict__ kR, const _Float16* __restrict__ vR,
    int ldq, int ldkv,
    const float* __restrict__ biasAll, const float* __restrict__ lam_ptr,
    _Float16* __restrict__ oT, _Float16* __restrict__ oR, int ldo)
{
    __shared__ _Float16 KsSw[2][4][32][24];      // [stream][cTile][lane][j]
    __shared__ _Float16 VsSw[2][2][2][32][24];   // [stream][kStep][dTile][lane][j]
    __shared__ _Float16 Ps[2][64][72];           // mixed probabilities, row-major

    const int bh = blockIdx.x;
    const int b = bh / HEADS, h = bh % HEADS;
    const int tid  = threadIdx.x;
    const int lane = tid & 31;
    const int w    = tid >> 5;
    const int l16  = lane & 15;
    const int hi   = lane >> 4;
    const int kb   = hi * 8;
    const size_t rowbase = (size_t)b * NSEQ;
    const int hoff = h * HDIM;

    // ---- stage K (vectorized swizzle) and V (scalar swizzle), zero-pad rows >= 49
    {
        const int row = tid >> 1, seg = tid & 1;   // seg = 16-half d-segment
        const size_t goff = (rowbase + row) * (size_t)ldkv + hoff + seg * 16;
#pragma unroll
        for (int s = 0; s < 2; ++s) {
            const _Float16* kp = s ? kR : kT;
            h8 v0, v1;
            if (row < NSEQ) { const h8* p = (const h8*)(kp + goff); v0 = p[0]; v1 = p[1]; }
            else            { v0 = hz8(); v1 = hz8(); }
            h8* dst = (h8*)&KsSw[s][row >> 4][seg * 16 + (row & 15)][0];
            dst[0] = v0; dst[1] = v1;
        }
        const int ks = row >> 5, vhi = (row >> 4) & 1, vj = row & 15;
#pragma unroll
        for (int s = 0; s < 2; ++s) {
            const _Float16* vp = s ? vR : vT;
            h8 v0, v1;
            if (row < NSEQ) { const h8* p = (const h8*)(vp + goff); v0 = p[0]; v1 = p[1]; }
            else            { v0 = hz8(); v1 = hz8(); }
            _Float16* dst = &VsSw[s][ks][seg][vhi * 16][vj];
#pragma unroll
            for (int i = 0; i < 8; ++i) {
                dst[i * 24]       = v0[i];
                dst[(8 + i) * 24] = v1[i];
            }
        }
    }
    __syncthreads();

    // ---- Q fragments (vector loads, scale folded in as packed f16 mul) ----
    const _Float16 sch = (_Float16)0.17677669529663687f;  // 32^-0.5
    v16h aT, aR;
    {
        const int arow = 16 * w + l16;
        if (arow < NSEQ) {
            const _Float16* pT = qT + (rowbase + arow) * ldq + hoff;
            const _Float16* pR = qR + (rowbase + arow) * ldq + hoff;
            v16h t = cat16(*(const h8*)(pT + hi * 16), *(const h8*)(pT + hi * 16 + 8));
            v16h r = cat16(*(const h8*)(pR + hi * 16), *(const h8*)(pR + hi * 16 + 8));
#pragma unroll
            for (int j = 0; j < 16; ++j) { aT[j] = t[j] * sch; aR[j] = r[j] * sch; }
        } else {
#pragma unroll
            for (int j = 0; j < 16; ++j) { aT[j] = (_Float16)0.0f; aR[j] = (_Float16)0.0f; }
        }
    }

    // ---- S = Q K^T for both streams ----
    v8f st[4], sr[4];
#pragma unroll
    for (int c = 0; c < 4; ++c) { st[c] = vz8(); sr[c] = vz8(); }
#pragma unroll
    for (int c = 0; c < 4; ++c) {
        h8 kt0 = *(const h8*)&KsSw[0][c][lane][0];
        h8 kt1 = *(const h8*)&KsSw[0][c][lane][8];
        h8 kr0 = *(const h8*)&KsSw[1][c][lane][0];
        h8 kr1 = *(const h8*)&KsSw[1][c][lane][8];
        st[c] = wmma_f16(aT, cat16(kt0, kt1), st[c]);
        sr[c] = wmma_f16(aR, cat16(kr0, kr1), sr[c]);
    }

    // ---- bias + dual softmax + lambda mix -> P1,P2 in LDS (f16) ----
    const float* bh_bias = biasAll + h * NSEQ * NSEQ;
    float lamv;
    {
        float lp = *lam_ptr;
        float s = 1.0f / (1.0f + __expf(-lp));
        lamv = fminf(0.99f, fmaxf(0.01f, s));
    }
#pragma unroll
    for (int r = 0; r < 8; ++r) {
        const int row = 16 * w + hi * 8 + r;
        const int rc = row < NSEQ ? row : NSEQ - 1;
        float mt = -1e30f, mr = -1e30f;
#pragma unroll
        for (int c = 0; c < 4; ++c) {
            const int col = c * 16 + l16;
            float vt_ = st[c][r], vr_ = sr[c][r];
            if (col < NSEQ) {
                float bb = bh_bias[rc * NSEQ + col];
                vt_ += bb; vr_ += bb;
            } else { vt_ = -1e30f; vr_ = -1e30f; }
            st[c][r] = vt_; sr[c][r] = vr_;
            mt = fmaxf(mt, vt_); mr = fmaxf(mr, vr_);
        }
#pragma unroll
        for (int m = 1; m < 16; m <<= 1) {
            mt = fmaxf(mt, __shfl_xor(mt, m, 32));
            mr = fmaxf(mr, __shfl_xor(mr, m, 32));
        }
        float et[4], er[4], sumt = 0.0f, sumr = 0.0f;
#pragma unroll
        for (int c = 0; c < 4; ++c) {
            et[c] = __expf(st[c][r] - mt);
            er[c] = __expf(sr[c][r] - mr);
            sumt += et[c]; sumr += er[c];
        }
#pragma unroll
        for (int m = 1; m < 16; m <<= 1) {
            sumt += __shfl_xor(sumt, m, 32);
            sumr += __shfl_xor(sumr, m, 32);
        }
        const float it = 1.0f / sumt, ir = 1.0f / sumr;
#pragma unroll
        for (int c = 0; c < 4; ++c) {
            const int col = c * 16 + l16;
            const float pt = et[c] * it, pr = er[c] * ir;
            Ps[0][row][col] = (_Float16)(pt - lamv * pr);
            Ps[1][row][col] = (_Float16)(pr - lamv * pt);
        }
    }
    __syncthreads();

    // ---- O = P @ V for both streams (K = 64 -> 2 WMMA steps; D = 32 -> 2 tiles)
    v8f o1[2], o2[2];
#pragma unroll
    for (int ct = 0; ct < 2; ++ct) { o1[ct] = vz8(); o2[ct] = vz8(); }
    const int prow = 16 * w + l16;
#pragma unroll
    for (int ks = 0; ks < 2; ++ks) {
        v16h a1 = cat16(*(const h8*)&Ps[0][prow][ks * 32 + hi * 16],
                        *(const h8*)&Ps[0][prow][ks * 32 + hi * 16 + 8]);
        v16h a2 = cat16(*(const h8*)&Ps[1][prow][ks * 32 + hi * 16],
                        *(const h8*)&Ps[1][prow][ks * 32 + hi * 16 + 8]);
#pragma unroll
        for (int ct = 0; ct < 2; ++ct) {
            h8 b10 = *(const h8*)&VsSw[0][ks][ct][lane][0];
            h8 b11 = *(const h8*)&VsSw[0][ks][ct][lane][8];
            h8 b20 = *(const h8*)&VsSw[1][ks][ct][lane][0];
            h8 b21 = *(const h8*)&VsSw[1][ks][ct][lane][8];
            o1[ct] = wmma_f16(a1, cat16(b10, b11), o1[ct]);
            o2[ct] = wmma_f16(a2, cat16(b20, b21), o2[ct]);
        }
    }

    // ---- store o_t / o_r (f16) ----
#pragma unroll
    for (int ct = 0; ct < 2; ++ct) {
#pragma unroll
        for (int r = 0; r < 8; ++r) {
            const int row = 16 * w + hi * 8 + r;
            if (row < NSEQ) {
                const int col = hoff + ct * 16 + l16;
                oT[(rowbase + row) * ldo + col] = (_Float16)o1[ct][r];
                oR[(rowbase + row) * ldo + col] = (_Float16)o2[ct][r];
            }
        }
    }
}

// ---------------------------------------------------------------------------
// Host orchestration
// ---------------------------------------------------------------------------
extern "C" void kernel_launch(void* const* d_in, const int* in_sizes, int n_in,
                              void* d_out, int out_size, void* d_ws, size_t ws_size,
                              hipStream_t stream) {
    (void)in_sizes; (void)n_in; (void)out_size; (void)ws_size;
    const float* x_sa      = (const float*)d_in[0];
    const float* x_ca      = (const float*)d_in[1];
    const float* lam_sa    = (const float*)d_in[2];
    const float* lam_ca    = (const float*)d_in[3];
    const float* sa_enh    = (const float*)d_in[4];
    const float* ca_enh    = (const float*)d_in[5];
    const float* W_sa_qkv  = (const float*)d_in[6];
    const float* b_sa_qkv  = (const float*)d_in[7];
    const float* W_sa_ct   = (const float*)d_in[8];
    const float* W_sa_cr   = (const float*)d_in[9];
    const float* W_ca_q    = (const float*)d_in[10];
    const float* b_ca_q    = (const float*)d_in[11];
    const float* W_ca_kv   = (const float*)d_in[12];
    const float* b_ca_kv   = (const float*)d_in[13];
    const float* W_ca_ct   = (const float*)d_in[14];
    const float* W_ca_cr   = (const float*)d_in[15];
    const float* rpb       = (const float*)d_in[16];
    const float* W_proj_sa = (const float*)d_in[17];
    const float* b_proj_sa = (const float*)d_in[18];
    const float* W_proj_ca = (const float*)d_in[19];
    const float* b_proj_ca = (const float*)d_in[20];
    float* out = (float*)d_out;

    const int Bb = 2048, Cc = CDIM, Hh = HEADS;
    const size_t M = (size_t)Bb * NSEQ;  // 100352 (multiple of 64)

    char* ws = (char*)d_ws;
    size_t off = 0;
    auto allocH = [&](size_t elems) {
        off = (off + 255) & ~(size_t)255;
        _Float16* p = (_Float16*)(ws + off); off += elems * sizeof(_Float16); return p;
    };
    auto allocF = [&](size_t elems) {
        off = (off + 255) & ~(size_t)255;
        float* p = (float*)(ws + off); off += elems * sizeof(float); return p;
    };

    // f16 weights
    _Float16* w_sa_cr_h   = allocH((size_t)Cc * Cc);
    _Float16* w_sa_ct_h   = allocH((size_t)Cc * Cc);
    _Float16* w_sa_qkv_h  = allocH((size_t)Cc * 3 * Cc);
    _Float16* w_ca_cr_h   = allocH((size_t)Cc * Cc);
    _Float16* w_ca_ct_h   = allocH((size_t)Cc * Cc);
    _Float16* w_ca_q_h    = allocH((size_t)Cc * Cc);
    _Float16* w_ca_kv_h   = allocH((size_t)Cc * 2 * Cc);
    _Float16* w_proj_sa_h = allocH((size_t)Cc * Cc);
    _Float16* w_proj_ca_h = allocH((size_t)Cc * Cc);
    float*    biasb       = allocF((size_t)Hh * NSEQ * NSEQ);
    // f16 activations (enhanced streams; reused as attention outputs later)
    _Float16* te_sa   = allocH(M * Cc);
    _Float16* re_sa   = allocH(M * Cc);
    _Float16* te_ca   = allocH(M * Cc);
    _Float16* re_ca   = allocH(M * Cc);
    _Float16* qkv_t   = allocH(M * 3 * Cc);
    _Float16* qkv_r   = allocH(M * 3 * Cc);
    _Float16* q_ca_t  = allocH(M * Cc);
    _Float16* q_ca_r  = allocH(M * Cc);
    _Float16* kv_te   = allocH(M * 2 * Cc);
    _Float16* kv_re   = allocH(M * 2 * Cc);
    // attention outputs alias te/re (dead after the QKV/Q/KV GEMMs)
    _Float16* o_t_sa = te_sa;
    _Float16* o_r_sa = re_sa;
    _Float16* o_t_ca = te_ca;
    _Float16* o_r_ca = re_ca;

    auto cvt = [&](const float* s, _Float16* d, size_t n) {
        unsigned blocks = (unsigned)((n + 511) / 512);
        cvt_f32_f16<<<blocks, 512, 0, stream>>>(s, d, (long long)n);
    };

    // 1) weight + bias prep (tiny)
    cvt(W_sa_cr,   w_sa_cr_h,   (size_t)Cc * Cc);
    cvt(W_sa_ct,   w_sa_ct_h,   (size_t)Cc * Cc);
    cvt(W_sa_qkv,  w_sa_qkv_h,  (size_t)Cc * 3 * Cc);
    cvt(W_ca_cr,   w_ca_cr_h,   (size_t)Cc * Cc);
    cvt(W_ca_ct,   w_ca_ct_h,   (size_t)Cc * Cc);
    cvt(W_ca_q,    w_ca_q_h,    (size_t)Cc * Cc);
    cvt(W_ca_kv,   w_ca_kv_h,   (size_t)Cc * 2 * Cc);
    cvt(W_proj_sa, w_proj_sa_h, (size_t)Cc * Cc);
    cvt(W_proj_ca, w_proj_ca_h, (size_t)Cc * Cc);
    build_bias_kernel<<<(Hh * NSEQ * NSEQ + 255) / 256, 256, 0, stream>>>(rpb, biasb);

    auto gemm = [&](const _Float16* Ah, const float* Af, int a_map,
                    const _Float16* Bw2, int Nout, const float* bias2,
                    const _Float16* addH, const float* addF, int add_map,
                    const float* alpha,
                    _Float16* oh, float* of, int out_map) {
        dim3 grid((unsigned)(Nout / 64), (unsigned)(M / 64));
        gemm_f16_wmma<<<grid, 128, 0, stream>>>(Ah, Af, Cc, a_map, Bw2, Nout, bias2,
                                                addH, addF, Cc, add_map, alpha,
                                                oh, Nout, of, Cc, out_map,
                                                (int)M, Nout, Cc);
    };

    // 2) cross-enhancement (A read directly as f32 from inputs, converted while
    //    staging; residual addend also read as f32 with CA row remap):
    //    te = xt + enh*(xr@W_cr),  re = xr + enh*(xt@W_ct)
    gemm(nullptr, x_sa + M * Cc, 0, w_sa_cr_h, Cc, nullptr, nullptr, x_sa,          0, sa_enh, te_sa, nullptr, 0);
    gemm(nullptr, x_sa,          0, w_sa_ct_h, Cc, nullptr, nullptr, x_sa + M * Cc, 0, sa_enh, re_sa, nullptr, 0);
    gemm(nullptr, x_ca,          2, w_ca_cr_h, Cc, nullptr, nullptr, x_ca,          1, ca_enh, te_ca, nullptr, 0);
    gemm(nullptr, x_ca,          1, w_ca_ct_h, Cc, nullptr, nullptr, x_ca,          2, ca_enh, re_ca, nullptr, 0);

    // 3) QKV / Q / KV projections (f16 -> f16)
    gemm(te_sa, nullptr, 0, w_sa_qkv_h, 3 * Cc, b_sa_qkv, nullptr, nullptr, 0, nullptr, qkv_t,  nullptr, 0);
    gemm(re_sa, nullptr, 0, w_sa_qkv_h, 3 * Cc, b_sa_qkv, nullptr, nullptr, 0, nullptr, qkv_r,  nullptr, 0);
    gemm(te_ca, nullptr, 0, w_ca_q_h,       Cc, b_ca_q,   nullptr, nullptr, 0, nullptr, q_ca_t, nullptr, 0);
    gemm(re_ca, nullptr, 0, w_ca_q_h,       Cc, b_ca_q,   nullptr, nullptr, 0, nullptr, q_ca_r, nullptr, 0);
    gemm(te_ca, nullptr, 0, w_ca_kv_h,  2 * Cc, b_ca_kv,  nullptr, nullptr, 0, nullptr, kv_te,  nullptr, 0);
    gemm(re_ca, nullptr, 0, w_ca_kv_h,  2 * Cc, b_ca_kv,  nullptr, nullptr, 0, nullptr, kv_re,  nullptr, 0);

    // 4) fused differential attention (one block per (b,h))
    diff_attn_kernel<<<Bb * Hh, 128, 0, stream>>>(
        qkv_t, qkv_t + Cc, qkv_t + 2 * Cc,
        qkv_r, qkv_r + Cc, qkv_r + 2 * Cc,
        3 * Cc, 3 * Cc, biasb, lam_sa, o_t_sa, o_r_sa, Cc);
    diff_attn_kernel<<<Bb * Hh, 128, 0, stream>>>(
        q_ca_t, kv_te, kv_te + Cc,
        q_ca_r, kv_re, kv_re + Cc,
        Cc, 2 * Cc, biasb, lam_ca, o_t_ca, o_r_ca, Cc);

    // 5) output projections straight into d_out (f32)
    // out shape (2, 2B, N, C): SA t rows [0,M), SA r rows [M,2M),
    // CA section at 2M with interleaved (2b / 2b+1) row mapping.
    gemm(o_t_sa, nullptr, 0, w_proj_sa_h, Cc, b_proj_sa, nullptr, nullptr, 0, nullptr, nullptr, out,              0);
    gemm(o_r_sa, nullptr, 0, w_proj_sa_h, Cc, b_proj_sa, nullptr, nullptr, 0, nullptr, nullptr, out + M * Cc,     0);
    gemm(o_t_ca, nullptr, 0, w_proj_ca_h, Cc, b_proj_ca, nullptr, nullptr, 0, nullptr, nullptr, out + 2 * M * Cc, 1);
    gemm(o_r_ca, nullptr, 0, w_proj_ca_h, Cc, b_proj_ca, nullptr, nullptr, 0, nullptr, nullptr, out + 2 * M * Cc, 2);
}
